// CLIPLoss_81063212745428
// MI455X (gfx1250) — matrix operand order repaired and checked by reading, not compiled, not run on hardware
//
#include <hip/hip_runtime.h>

// ---------------- types ----------------
typedef __attribute__((ext_vector_type(16))) __bf16 v16bf;
typedef __attribute__((ext_vector_type(8)))  __bf16 v8bf;
typedef __attribute__((ext_vector_type(8)))  float  v8f;
typedef __attribute__((ext_vector_type(4)))  int    v4i;

#define BB 16384
#define DD 768
#define LDA 40                 // padded LDS row stride (elements): 80B = 20 banks, conflict-free
#define MATSZ (128 * LDA)      // one staged matrix buffer (elements)

#define GLOBAL_AS __attribute__((address_space(1)))
#define LDS_AS    __attribute__((address_space(3)))

#if defined(__gfx1250__) && __has_builtin(__builtin_amdgcn_global_load_async_to_lds_b128)
  #define HAVE_ASYNC 1
#else
  #define HAVE_ASYNC 0
#endif

#if HAVE_ASYNC
  #if __has_builtin(__builtin_amdgcn_s_wait_asynccnt)
    #define WAIT_ASYNC(n) __builtin_amdgcn_s_wait_asynccnt(n)
  #else
    #define WAIT_ASYNC(n) asm volatile("s_wait_asynccnt %0" :: "i"(n) : "memory")
  #endif
#else
  #define WAIT_ASYNC(n) ((void)0)
#endif

#define WMMA_BF16(a, b, c) \
  __builtin_amdgcn_wmma_f32_16x16x32_bf16(false, (a), false, (b), (short)0, (c), false, false)

// ---------------- helpers ----------------
__device__ __forceinline__ unsigned short f2bf(float f) {
    unsigned u = __float_as_uint(f);
    u += 0x7FFFu + ((u >> 16) & 1u);      // round-to-nearest-even
    return (unsigned short)(u >> 16);
}
// order-preserving float <-> uint key (monotone under unsigned compare)
__device__ __forceinline__ unsigned f2key(float f) {
    unsigned u = __float_as_uint(f);
    return (u & 0x80000000u) ? ~u : (u | 0x80000000u);
}
__device__ __forceinline__ float key2f(unsigned k) {
    unsigned u = (k & 0x80000000u) ? (k ^ 0x80000000u) : ~k;
    return __uint_as_float(u);
}
__device__ __forceinline__ float red_max16(float v) {
    #pragma unroll
    for (int m = 1; m < 16; m <<= 1) v = fmaxf(v, __shfl_xor(v, m, 32));
    return v;   // stays within 16-lane halves of the wave32
}
__device__ __forceinline__ float red_sum16(float v) {
    #pragma unroll
    for (int m = 1; m < 16; m <<= 1) v += __shfl_xor(v, m, 32);
    return v;
}

// ---------------- LDS fragment loads (ISA 16-bit A/B layouts) ----------------
// A 16x32: lane L: row=row0+(L&15); K chunks {h8..h8+7} and {h8+16..h8+23}, h8=8*(L>=16)
__device__ __forceinline__ v16bf lds_frag_a(const unsigned short* s, int row0, int lane) {
    const unsigned short* p = s + (row0 + (lane & 15)) * LDA + ((lane >> 4) << 3);
    union { v16bf v; v8bf h[2]; } u;
    u.h[0] = *(const v8bf*)(p);
    u.h[1] = *(const v8bf*)(p + 16);
    return u.v;
}
// B 32x16: lane L: column row0+(L&15) (a row of txt); K chunks {h8..h8+7} and {h8+16..h8+23}, h8=8*(L>=16)
__device__ __forceinline__ v16bf lds_frag_b(const unsigned short* s, int row0, int lane) {
    const unsigned short* p = s + (row0 + (lane & 15)) * LDA + ((lane >> 4) << 3);
    union { v16bf v; v8bf h[2]; } u;
    u.h[0] = *(const v8bf*)(p);
    u.h[1] = *(const v8bf*)(p + 16);
    return u.v;
}

// ---------------- cooperative stage copy: 128x32 of A and of B -> LDS ----------------
// 256 threads; each thread moves 4x16B (2 for A, 2 for B) => ASYNCcnt += 4 per wave
__device__ __forceinline__ void stage(const unsigned short* __restrict__ gA,
                                      const unsigned short* __restrict__ gB,
                                      unsigned short* smem, int buf,
                                      int mrow0, int nrow0, int k, int t) {
    const int part = t & 3;          // 16B chunk within a 64B row
    const int row  = t >> 2;         // 0..63
    #pragma unroll
    for (int p = 0; p < 2; p++) {
        const int r = row + p * 64;
        const size_t goa = (size_t)(mrow0 + r) * DD + k + part * 8;
        const size_t gob = (size_t)(nrow0 + r) * DD + k + part * 8;
        unsigned short* la = smem + buf * MATSZ       + r * LDA + part * 8;
        unsigned short* lb = smem + (2 + buf) * MATSZ + r * LDA + part * 8;
#if HAVE_ASYNC
        __builtin_amdgcn_global_load_async_to_lds_b128(
            (GLOBAL_AS v4i*)(gA + goa), (LDS_AS v4i*)la, 0, 0);
        __builtin_amdgcn_global_load_async_to_lds_b128(
            (GLOBAL_AS v4i*)(gB + gob), (LDS_AS v4i*)lb, 0, 0);
#else
        *(uint4*)la = *(const uint4*)(gA + goa);
        *(uint4*)lb = *(const uint4*)(gB + gob);
#endif
    }
}

// ---------------- 128x128 block GEMM, wave tile 32x64, double-buffered LDS ----------------
__device__ __forceinline__ void gemm_block(const unsigned short* __restrict__ imgb,
                                           const unsigned short* __restrict__ txtb,
                                           int mblk0, int nblk0, float scale,
                                           v8f (&c)[2][4]) {
    __shared__ unsigned short smem[4 * MATSZ];   // A0,A1,B0,B1 = 40KB
    const int t    = threadIdx.x;
    const int lane = t & 31;
    const int w    = t >> 5;
    const int wm   = w >> 1;   // 0..3 -> M offset wm*32
    const int wn   = w & 1;    // 0..1 -> N offset wn*64

    #pragma unroll
    for (int i = 0; i < 2; i++)
        #pragma unroll
        for (int j = 0; j < 4; j++) c[i][j] = v8f{};

    stage(imgb, txtb, smem, 0, mblk0, nblk0, 0, t);
    for (int ks = 0; ks < DD / 32; ks++) {
        const int buf = ks & 1;
        if (ks + 1 < DD / 32) {
            stage(imgb, txtb, smem, buf ^ 1, mblk0, nblk0, (ks + 1) * 32, t);
            WAIT_ASYNC(4);   // in-order: current stage's 4 copies done; next 4 may pend
        } else {
            WAIT_ASYNC(0);
        }
        __syncthreads();
        const unsigned short* sA = smem + buf * MATSZ;
        const unsigned short* sB = smem + (2 + buf) * MATSZ;
        v16bf a0 = lds_frag_a(sA, wm * 32,      lane);
        v16bf a1 = lds_frag_a(sA, wm * 32 + 16, lane);
        v16bf b0 = lds_frag_b(sB, wn * 64,      lane);
        v16bf b1 = lds_frag_b(sB, wn * 64 + 16, lane);
        v16bf b2 = lds_frag_b(sB, wn * 64 + 32, lane);
        v16bf b3 = lds_frag_b(sB, wn * 64 + 48, lane);
        c[0][0] = WMMA_BF16(a0, b0, c[0][0]);
        c[0][1] = WMMA_BF16(a0, b1, c[0][1]);
        c[0][2] = WMMA_BF16(a0, b2, c[0][2]);
        c[0][3] = WMMA_BF16(a0, b3, c[0][3]);
        c[1][0] = WMMA_BF16(a1, b0, c[1][0]);
        c[1][1] = WMMA_BF16(a1, b1, c[1][1]);
        c[1][2] = WMMA_BF16(a1, b2, c[1][2]);
        c[1][3] = WMMA_BF16(a1, b3, c[1][3]);
        __syncthreads();
    }
    #pragma unroll
    for (int i = 0; i < 2; i++)
        #pragma unroll
        for (int j = 0; j < 4; j++)
            #pragma unroll
            for (int r = 0; r < 8; r++) c[i][j][r] *= scale;
}

// ---------------- kernel 1: convert + init ----------------
__global__ __launch_bounds__(256) void clip_convert(
    const float* __restrict__ img, const float* __restrict__ txt,
    unsigned short* __restrict__ imgb, unsigned short* __restrict__ txtb,
    unsigned* __restrict__ rkey, unsigned* __restrict__ ckey,
    float* __restrict__ rsum, float* __restrict__ csum) {
    const size_t idx = (size_t)blockIdx.x * 256 + threadIdx.x;
    const size_t total = (size_t)BB * DD;
    if (idx < total) {
        imgb[idx] = f2bf(img[idx]);
        txtb[idx] = f2bf(txt[idx]);
    }
    if (idx < BB) {
        rkey[idx] = 0u; ckey[idx] = 0u;
        rsum[idx] = 0.0f; csum[idx] = 0.0f;
    }
}

// ---------------- kernel 2: GEMM + global row/col max ----------------
__global__ __launch_bounds__(256) void clip_gemm_max(
    const unsigned short* __restrict__ imgb, const unsigned short* __restrict__ txtb,
    const float* __restrict__ p_ls,
    unsigned* __restrict__ rkey, unsigned* __restrict__ ckey) {
    const int lane = threadIdx.x & 31;
    const int w    = threadIdx.x >> 5;
    const int mblk0 = blockIdx.y * 128, nblk0 = blockIdx.x * 128;
    const int m_base = mblk0 + (w >> 1) * 32;
    const int n_base = nblk0 + (w & 1) * 64;
    const float scale = fminf(expf(*p_ls), 100.0f);

    v8f c[2][4];
    gemm_block(imgb, txtb, mblk0, nblk0, scale, c);

    const int h8 = (lane >> 4) << 3;
    // row maxima: combine the 4 N-subtiles first, then one 16-lane reduction, 1 atomic/row
    #pragma unroll
    for (int r = 0; r < 8; r++) {
        float rm0 = c[0][0][r], rm1 = c[1][0][r];
        #pragma unroll
        for (int j = 1; j < 4; j++) {
            rm0 = fmaxf(rm0, c[0][j][r]);
            rm1 = fmaxf(rm1, c[1][j][r]);
        }
        rm0 = red_max16(rm0);
        rm1 = red_max16(rm1);
        if ((lane & 15) == 0) {
            atomicMax(&rkey[m_base + r + h8],      f2key(rm0));
            atomicMax(&rkey[m_base + 16 + r + h8], f2key(rm1));
        }
    }
    // column maxima: reduce over the 16 C VGPRs per subtile, fold halves, 2 atomics/lane
    float cm[4];
    #pragma unroll
    for (int j = 0; j < 4; j++) {
        float v = fmaxf(c[0][j][0], c[1][j][0]);
        #pragma unroll
        for (int r = 1; r < 8; r++) v = fmaxf(v, fmaxf(c[0][j][r], c[1][j][r]));
        cm[j] = fmaxf(v, __shfl_xor(v, 16, 32));
    }
    const int j0 = lane >> 4;
    atomicMax(&ckey[n_base + j0 * 16 + (lane & 15)],       f2key(cm[j0]));
    atomicMax(&ckey[n_base + (j0 + 2) * 16 + (lane & 15)], f2key(cm[j0 + 2]));
}

// ---------------- kernel 3: GEMM + exp-sums + diagonal ----------------
__global__ __launch_bounds__(256) void clip_gemm_sum(
    const unsigned short* __restrict__ imgb, const unsigned short* __restrict__ txtb,
    const float* __restrict__ p_ls,
    const unsigned* __restrict__ rkey, const unsigned* __restrict__ ckey,
    float* __restrict__ rsum, float* __restrict__ csum, float* __restrict__ diag) {
    const int lane = threadIdx.x & 31;
    const int w    = threadIdx.x >> 5;
    const int mblk0 = blockIdx.y * 128, nblk0 = blockIdx.x * 128;
    const int m_base = mblk0 + (w >> 1) * 32;
    const int n_base = nblk0 + (w & 1) * 64;
    const float scale = fminf(expf(*p_ls), 100.0f);

    v8f c[2][4];
    gemm_block(imgb, txtb, mblk0, nblk0, scale, c);

    const int h8 = (lane >> 4) << 3;
    float rmv0[8], rmv1[8];
    #pragma unroll
    for (int r = 0; r < 8; r++) {
        rmv0[r] = key2f(rkey[m_base + r + h8]);
        rmv1[r] = key2f(rkey[m_base + 16 + r + h8]);
    }
    float cmn[4], cs[4];
    #pragma unroll
    for (int j = 0; j < 4; j++) {
        cmn[j] = key2f(ckey[n_base + j * 16 + (lane & 15)]);
        cs[j] = 0.0f;
    }

    #pragma unroll
    for (int r = 0; r < 8; r++) {
        const int gm0 = m_base + r + h8;
        const int gm1 = gm0 + 16;
        float rs0 = 0.0f, rs1 = 0.0f;
        #pragma unroll
        for (int j = 0; j < 4; j++) {
            const float l0 = c[0][j][r], l1 = c[1][j][r];
            const int gn = n_base + j * 16 + (lane & 15);
            if (gm0 == gn) diag[gm0] = l0;   // each diagonal element hit exactly once
            if (gm1 == gn) diag[gm1] = l1;
            rs0 += expf(l0 - rmv0[r]);
            rs1 += expf(l1 - rmv1[r]);
            cs[j] += expf(l0 - cmn[j]) + expf(l1 - cmn[j]);
        }
        rs0 = red_sum16(rs0);
        rs1 = red_sum16(rs1);
        if ((lane & 15) == 0) {
            atomicAdd(&rsum[gm0], rs0);
            atomicAdd(&rsum[gm1], rs1);
        }
    }
    #pragma unroll
    for (int j = 0; j < 4; j++) cs[j] += __shfl_xor(cs[j], 16, 32);
    const int j0 = lane >> 4;
    atomicAdd(&csum[n_base + j0 * 16 + (lane & 15)],       cs[j0]);
    atomicAdd(&csum[n_base + (j0 + 2) * 16 + (lane & 15)], cs[j0 + 2]);
}

// ---------------- kernel 4: finalize ----------------
__global__ __launch_bounds__(256) void clip_finalize(
    const unsigned* __restrict__ rkey, const unsigned* __restrict__ ckey,
    const float* __restrict__ rsum, const float* __restrict__ csum,
    const float* __restrict__ diag, float* __restrict__ out) {
    __shared__ float s0[256], s1[256], s2[256], s3[256];
    const int tid = threadIdx.x;
    float li = 0.0f, lt = 0.0f, ai = 0.0f, at = 0.0f;
    for (int i = tid; i < BB; i += 256) {
        const float rm = key2f(rkey[i]);
        const float cm = key2f(ckey[i]);
        const float d  = diag[i];
        li += (rm + logf(rsum[i])) - d;   // -log p_i2t[i,i]
        lt += (cm + logf(csum[i])) - d;   // -log p_t2i[i,i]
        ai += (d >= rm) ? 1.0f : 0.0f;    // diag is row max <=> argmax == label
        at += (d >= cm) ? 1.0f : 0.0f;
    }
    s0[tid] = li; s1[tid] = lt; s2[tid] = ai; s3[tid] = at;
    __syncthreads();
    for (int s = 128; s > 0; s >>= 1) {
        if (tid < s) {
            s0[tid] += s0[tid + s]; s1[tid] += s1[tid + s];
            s2[tid] += s2[tid + s]; s3[tid] += s3[tid + s];
        }
        __syncthreads();
    }
    if (tid == 0) {
        const float inv = 1.0f / (float)BB;
        out[0] = 0.5f * (s0[0] + s1[0]) * inv;  // loss
        out[1] = s2[0] * inv;                    // i2t accuracy
        out[2] = s3[0] * inv;                    // t2i accuracy
    }
}

// ---------------- launcher ----------------
extern "C" void kernel_launch(void* const* d_in, const int* in_sizes, int n_in,
                              void* d_out, int out_size, void* d_ws, size_t ws_size,
                              hipStream_t stream) {
    const float* img = (const float*)d_in[0];
    const float* txt = (const float*)d_in[1];
    const float* ls  = (const float*)d_in[2];
    float* out = (float*)d_out;

    const size_t ND = (size_t)BB * DD;
    unsigned short* imgb = (unsigned short*)d_ws;
    unsigned short* txtb = imgb + ND;
    unsigned* rkey = (unsigned*)(txtb + ND);
    unsigned* ckey = rkey + BB;
    float* rsum = (float*)(ckey + BB);
    float* csum = rsum + BB;
    float* diag = csum + BB;

    const int cvt_blocks = (int)((ND + 255) / 256);
    clip_convert<<<cvt_blocks, 256, 0, stream>>>(img, txt, imgb, txtb, rkey, ckey, rsum, csum);

    dim3 g(BB / 128, BB / 128);
    clip_gemm_max<<<g, 256, 0, stream>>>(imgb, txtb, ls, rkey, ckey);
    clip_gemm_sum<<<g, 256, 0, stream>>>(imgb, txtb, ls, rkey, ckey, rsum, csum, diag);

    clip_finalize<<<1, 256, 0, stream>>>(rkey, ckey, rsum, csum, diag, out);
}
